// GAT3_13838384627835
// MI455X (gfx1250) — matrix operand order repaired and checked by reading, not compiled, not run on hardware
//
#include <hip/hip_runtime.h>

typedef __attribute__((ext_vector_type(16))) __bf16 v16bf;
typedef __attribute__((ext_vector_type(4)))  __bf16 v4bf;
typedef __attribute__((ext_vector_type(8)))  float  v8f;

// ---------- helpers ----------
static __device__ __forceinline__ __bf16 f2bf(float f) {
    unsigned u = __float_as_uint(f);
    u = (u + 0x7FFFu + ((u >> 16) & 1u)) >> 16;   // round-to-nearest-even
    unsigned short h = (unsigned short)u;
    __bf16 b;
    __builtin_memcpy(&b, &h, 2);
    return b;
}

static __device__ __forceinline__ void atomicMaxF(float* addr, float val) {
    int* ia = (int*)addr;
    int cur = __float_as_int(*addr);
    while (val > __int_as_float(cur)) {
        int old = atomicCAS(ia, cur, __float_as_int(val));
        if (old == cur) break;
        cur = old;
    }
}

// ---------- fp32 -> bf16 bulk convert (n must be multiple of 4) ----------
__global__ __launch_bounds__(256)
void cvt_f32_to_bf16(const float* __restrict__ src, __bf16* __restrict__ dst, int n4) {
    int i = blockIdx.x * blockDim.x + threadIdx.x;
    if (i >= n4) return;
    float4 v = ((const float4*)src)[i];
    v4bf o;
    o[0] = f2bf(v.x); o[1] = f2bf(v.y); o[2] = f2bf(v.z); o[3] = f2bf(v.w);
    ((v4bf*)dst)[i] = o;
}

// ---------- WMMA GEMM: C[M,NC] = A[M,K](bf16) x W[K,NC](fp32, re-staged to LDS fragments) ----------
// 256 threads = 8 waves; wave w owns M-tile (blockIdx.x*8 + w).
// W is pre-shuffled into fragment-major LDS: [nt][kb32][lane][16 bf16] (32B contiguous per lane),
// so each B operand is two ds_load_b128; each A operand is two global_load_b128.
template<int K, int NC>
__global__ __launch_bounds__(256)
void gemm_wmma_bf16(const __bf16* __restrict__ Abf, const float* __restrict__ Wfp,
                    float* __restrict__ C, int Mtiles) {
    constexpr int KB = K / 32;     // k-steps
    constexpr int NT = NC / 16;    // n-tiles
    __shared__ __bf16 sFrag[K * NC];   // <= 64 KB (K=512, NC=64)

    // Cooperative fragment-major staging of W (bf16).
    // Fragment layout: lane l: j<8 -> K = kb + (l>=16?8:0) + j, j>=8 -> K = kb + 16 + (l>=16?8:0) + (j-8):
    //   lanes 0-15 hold K = kb+0..7,16..23, lanes 16-31 hold K = kb+8..15,24..31, column = nt*16 + (l&15).
    for (int idx = threadIdx.x; idx < K * NC; idx += 256) {
        int j    = idx & 15;
        int l    = (idx >> 4) & 31;
        int fg   = idx >> 9;            // fragment id = nt*KB + kb32
        int kb32 = fg % KB;
        int nt   = fg / KB;
        int gk   = kb32 * 32 + (j < 8 ? (l >> 4) * 8 + j : 16 + (l >> 4) * 8 + (j - 8));
        int col  = nt * 16 + (l & 15);
        sFrag[idx] = f2bf(Wfp[gk * NC + col]);
    }
    __syncthreads();

    const int lane = threadIdx.x & 31;
    const int wave = threadIdx.x >> 5;
    const int mt   = blockIdx.x * 8 + wave;
    if (mt >= Mtiles) return;

    const int r    = lane & 15;
    const int half = lane >> 4;
    const __bf16* __restrict__ arow = Abf + (size_t)(mt * 16 + r) * K;

    if constexpr ((NT & 1) == 0) {
        // paired N-tiles: share the A fragment across two independent accumulator chains
        for (int nt = 0; nt < NT; nt += 2) {
            v8f acc0 = {}, acc1 = {};
            #pragma unroll 4
            for (int kb32 = 0; kb32 < KB; ++kb32) {
                const int kb = kb32 * 32;
                // A fragment (ISA 16-bit A 16x32): lanes 0-15: K=kb+0..7,kb+16..23 ; lanes 16-31: +8
                v16bf a;
                __builtin_memcpy(&a, arow + kb + half * 8, 16);
                __builtin_memcpy((char*)&a + 16, arow + kb + 16 + half * 8, 16);
                v16bf b0, b1;
                __builtin_memcpy(&b0, sFrag + ((size_t)((nt + 0) * KB + kb32) * 32 + lane) * 16, 32);
                __builtin_memcpy(&b1, sFrag + ((size_t)((nt + 1) * KB + kb32) * 32 + lane) * 16, 32);
                acc0 = __builtin_amdgcn_wmma_f32_16x16x32_bf16(false, a, false, b0,
                                                               (short)0, acc0, false, false);
                acc1 = __builtin_amdgcn_wmma_f32_16x16x32_bf16(false, a, false, b1,
                                                               (short)0, acc1, false, false);
            }
            float* crow0 = C + (size_t)mt * 16 * NC + (nt + 0) * 16;
            float* crow1 = C + (size_t)mt * 16 * NC + (nt + 1) * 16;
            #pragma unroll
            for (int v = 0; v < 8; ++v) {
                crow0[(size_t)(v + half * 8) * NC + r] = acc0[v];
                crow1[(size_t)(v + half * 8) * NC + r] = acc1[v];
            }
        }
    } else {
        for (int nt = 0; nt < NT; ++nt) {
            v8f acc = {};
            #pragma unroll 4
            for (int kb32 = 0; kb32 < KB; ++kb32) {
                const int kb = kb32 * 32;
                v16bf a;
                __builtin_memcpy(&a, arow + kb + half * 8, 16);
                __builtin_memcpy((char*)&a + 16, arow + kb + 16 + half * 8, 16);
                v16bf b;
                __builtin_memcpy(&b, sFrag + ((size_t)(nt * KB + kb32) * 32 + lane) * 16, 32);
                acc = __builtin_amdgcn_wmma_f32_16x16x32_bf16(false, a, false, b,
                                                              (short)0, acc, false, false);
            }
            float* crow = C + (size_t)mt * 16 * NC + nt * 16;
            #pragma unroll
            for (int v = 0; v < 8; ++v)
                crow[(size_t)(v + half * 8) * NC + r] = acc[v];
        }
    }
}

// ---------- attention coefficients ----------
__global__ void coef_kernel(const float* __restrict__ Hb, const float* __restrict__ a_src,
                            const float* __restrict__ a_dst, float* __restrict__ es,
                            float* __restrict__ ed, int N, int Hh, int Cc) {
    int i = blockIdx.x * blockDim.x + threadIdx.x;
    if (i >= N * Hh) return;
    int h = i % Hh;
    const float* hv = Hb + (size_t)i * Cc;
    float s0 = 0.f, s1 = 0.f;
    for (int c = 0; c < Cc; ++c) {
        float v = hv[c];
        s0 += v * a_src[h * Cc + c];
        s1 += v * a_dst[h * Cc + c];
    }
    es[i] = s0; ed[i] = s1;
}

__global__ void fill_kernel(float* __restrict__ p, float v, int n) {
    int i = blockIdx.x * blockDim.x + threadIdx.x;
    if (i < n) p[i] = v;
}

// ---------- edge passes (edges: e<E -> edge_index, e>=E -> self loop e-E) ----------
__global__ void edge_passA(const int* __restrict__ ei, int E, int ET,
                           const float* __restrict__ es, const float* __restrict__ ed,
                           float* __restrict__ tmp, float* __restrict__ m, int Hh) {
    int i = blockIdx.x * blockDim.x + threadIdx.x;
    if (i >= ET * Hh) return;
    int h = i % Hh, e = i / Hh;
    int sn = (e < E) ? ei[e]     : (e - E);
    int dn = (e < E) ? ei[E + e] : (e - E);
    float x = es[sn * Hh + h] + ed[dn * Hh + h];
    x = (x > 0.f) ? x : 0.2f * x;                      // leaky_relu(0.2)
    tmp[i] = x;
    atomicMaxF(&m[dn * Hh + h], x);
}

__global__ void edge_passB(const int* __restrict__ ei, int E, int ET,
                           float* __restrict__ tmp, const float* __restrict__ m,
                           float* __restrict__ s, int Hh) {
    int i = blockIdx.x * blockDim.x + threadIdx.x;
    if (i >= ET * Hh) return;
    int h = i % Hh, e = i / Hh;
    int dn = (e < E) ? ei[E + e] : (e - E);
    float w = expf(tmp[i] - m[dn * Hh + h]);
    tmp[i] = w;
    atomicAdd(&s[dn * Hh + h], w);
}

__global__ void edge_passC(const int* __restrict__ ei, int E, int ET,
                           const float* __restrict__ tmp, const float* __restrict__ s,
                           const float* __restrict__ Hb, float* __restrict__ O,
                           int Hh, int Cc) {
    int i = blockIdx.x * blockDim.x + threadIdx.x;
    if (i >= ET * Hh) return;
    int h = i % Hh, e = i / Hh;
    int sn = (e < E) ? ei[e]     : (e - E);
    int dn = (e < E) ? ei[E + e] : (e - E);
    float alpha = tmp[i] / s[dn * Hh + h];
    const float* hs = Hb + ((size_t)sn * Hh + h) * Cc;
    float* od       = O  + ((size_t)dn * Hh + h) * Cc;
    for (int c = 0; c < Cc; c += 4) {
        float4 v = *(const float4*)(hs + c);
        atomicAdd(od + c + 0, alpha * v.x);
        atomicAdd(od + c + 1, alpha * v.y);
        atomicAdd(od + c + 2, alpha * v.z);
        atomicAdd(od + c + 3, alpha * v.w);
    }
}

// ---------- per-layer finalize ----------
__global__ void fin_concat_elu(const float* __restrict__ O, const float* __restrict__ bias,
                               float* __restrict__ A, int N, int F) {
    int i = blockIdx.x * blockDim.x + threadIdx.x;
    if (i >= N * F) return;
    float v = O[i] + bias[i % F];
    A[i] = (v > 0.f) ? v : (expf(v) - 1.f);
}

__global__ void fin_mean_elu(const float* __restrict__ O, const float* __restrict__ bias,
                             float* __restrict__ A, int N, int Hh, int Cc) {
    int i = blockIdx.x * blockDim.x + threadIdx.x;
    if (i >= N * Cc) return;
    int n = i / Cc, c = i % Cc;
    float acc = 0.f;
    for (int h = 0; h < Hh; ++h) acc += O[(size_t)n * Hh * Cc + h * Cc + c];
    float v = acc / (float)Hh + bias[c];
    A[i] = (v > 0.f) ? v : (expf(v) - 1.f);
}

__global__ void fin_pool(const float* __restrict__ O, const float* __restrict__ bias,
                         const int* __restrict__ batch, float* __restrict__ pool,
                         int N, int F) {
    int i = blockIdx.x * blockDim.x + threadIdx.x;
    if (i >= N * F) return;
    int n = i / F, c = i % F;
    atomicAdd(&pool[batch[n] * F + c], O[i] + bias[c]);
}

__global__ void count_kernel(const int* __restrict__ batch, float* __restrict__ cnt, int N) {
    int i = blockIdx.x * blockDim.x + threadIdx.x;
    if (i < N) atomicAdd(&cnt[batch[i]], 1.f);
}

__global__ void final_linear(const float* __restrict__ pool, const float* __restrict__ cnt,
                             const float* __restrict__ Wl, const float* __restrict__ bl,
                             float* __restrict__ out, int G) {
    int t = blockIdx.x * blockDim.x + threadIdx.x;
    if (t >= G * 2) return;
    int g = t >> 1, o = t & 1;
    float inv = 1.f / fmaxf(cnt[g], 1.f);
    float acc = bl[o];
    for (int k = 0; k < 16; ++k) acc += pool[g * 16 + k] * inv * Wl[k * 2 + o];
    out[t] = acc;
}

// ---------- host launch ----------
static inline int cdiv(long long a, long long b) { return (int)((a + b - 1) / b); }

extern "C" void kernel_launch(void* const* d_in, const int* in_sizes, int n_in,
                              void* d_out, int out_size, void* d_ws, size_t ws_size,
                              hipStream_t stream) {
    const float* x   = (const float*)d_in[0];
    const int*   ei  = (const int*)d_in[1];
    const int*   bat = (const int*)d_in[2];
    const float* W1  = (const float*)d_in[3];
    const float* as1 = (const float*)d_in[4];
    const float* ad1 = (const float*)d_in[5];
    const float* b1  = (const float*)d_in[6];
    const float* W2  = (const float*)d_in[7];
    const float* as2 = (const float*)d_in[8];
    const float* ad2 = (const float*)d_in[9];
    const float* b2  = (const float*)d_in[10];
    const float* W3  = (const float*)d_in[11];
    const float* as3 = (const float*)d_in[12];
    const float* ad3 = (const float*)d_in[13];
    const float* b3  = (const float*)d_in[14];
    const float* Wl  = (const float*)d_in[15];
    const float* bl  = (const float*)d_in[16];

    const int N  = in_sizes[0] / 512;
    const int E  = in_sizes[1] / 2;
    const int ET = E + N;
    const int G  = out_size / 2;
    const int Mtiles = cdiv(N, 16);

    float* ws   = (float*)d_ws;
    float* Abuf = ws;  ws += (size_t)N * 64;     // layer activations (fp32)
    float* Hb   = ws;  ws += (size_t)N * 256;    // GEMM output h
    float* O    = ws;  ws += (size_t)N * 256;    // aggregation accumulator
    float* es   = ws;  ws += (size_t)N * 8;
    float* ed   = ws;  ws += (size_t)N * 8;
    float* mb   = ws;  ws += (size_t)N * 8;
    float* sb   = ws;  ws += (size_t)N * 8;
    float* tmp  = ws;  ws += (size_t)ET * 8;
    float* pool = ws;  ws += (size_t)G * 16;
    float* cnt  = ws;  ws += G;
    // align to 16 bytes for bf16 region
    ws += (16 - ((uintptr_t)ws & 15)) & 15;      // (always 4-byte aligned floats; pad)
    __bf16* Abf = (__bf16*)ws;                    // N*512 bf16 = N*256 floats
    ws += (size_t)N * 256;

    const int TB = 256;

    // ================= layer 1: 512 -> 8 heads x 8, concat =================
    {
        const int Hh = 8, Cc = 8, F = 64;
        cvt_f32_to_bf16<<<cdiv((long long)N * 512 / 4, TB), TB, 0, stream>>>(x, Abf, N * 512 / 4);
        gemm_wmma_bf16<512, 64><<<cdiv(Mtiles, 8), TB, 0, stream>>>(Abf, W1, Hb, Mtiles);
        coef_kernel<<<cdiv((long long)N * Hh, TB), TB, 0, stream>>>(Hb, as1, ad1, es, ed, N, Hh, Cc);
        fill_kernel<<<cdiv((long long)N * Hh, TB), TB, 0, stream>>>(mb, -1e30f, N * Hh);
        fill_kernel<<<cdiv((long long)N * Hh, TB), TB, 0, stream>>>(sb, 0.f, N * Hh);
        fill_kernel<<<cdiv((long long)N * F, TB), TB, 0, stream>>>(O, 0.f, N * F);
        int EH = ET * Hh;
        edge_passA<<<cdiv(EH, TB), TB, 0, stream>>>(ei, E, ET, es, ed, tmp, mb, Hh);
        edge_passB<<<cdiv(EH, TB), TB, 0, stream>>>(ei, E, ET, tmp, mb, sb, Hh);
        edge_passC<<<cdiv(EH, TB), TB, 0, stream>>>(ei, E, ET, tmp, sb, Hb, O, Hh, Cc);
        fin_concat_elu<<<cdiv((long long)N * F, TB), TB, 0, stream>>>(O, b1, Abuf, N, F);
    }
    // ================= layer 2: 64 -> 8 heads x 32, mean =================
    {
        const int Hh = 8, Cc = 32, F = 256;
        cvt_f32_to_bf16<<<cdiv((long long)N * 64 / 4, TB), TB, 0, stream>>>(Abuf, Abf, N * 64 / 4);
        gemm_wmma_bf16<64, 256><<<cdiv(Mtiles, 8), TB, 0, stream>>>(Abf, W2, Hb, Mtiles);
        coef_kernel<<<cdiv((long long)N * Hh, TB), TB, 0, stream>>>(Hb, as2, ad2, es, ed, N, Hh, Cc);
        fill_kernel<<<cdiv((long long)N * Hh, TB), TB, 0, stream>>>(mb, -1e30f, N * Hh);
        fill_kernel<<<cdiv((long long)N * Hh, TB), TB, 0, stream>>>(sb, 0.f, N * Hh);
        fill_kernel<<<cdiv((long long)N * F, TB), TB, 0, stream>>>(O, 0.f, N * F);
        int EH = ET * Hh;
        edge_passA<<<cdiv(EH, TB), TB, 0, stream>>>(ei, E, ET, es, ed, tmp, mb, Hh);
        edge_passB<<<cdiv(EH, TB), TB, 0, stream>>>(ei, E, ET, tmp, mb, sb, Hh);
        edge_passC<<<cdiv(EH, TB), TB, 0, stream>>>(ei, E, ET, tmp, sb, Hb, O, Hh, Cc);
        fin_mean_elu<<<cdiv((long long)N * Cc, TB), TB, 0, stream>>>(O, b2, Abuf, N, Hh, Cc);
    }
    // ================= layer 3: 32 -> 1 head x 16, concat; then pool + linear =================
    {
        const int Hh = 1, Cc = 16, F = 16;
        cvt_f32_to_bf16<<<cdiv((long long)N * 32 / 4, TB), TB, 0, stream>>>(Abuf, Abf, N * 32 / 4);
        gemm_wmma_bf16<32, 16><<<cdiv(Mtiles, 8), TB, 0, stream>>>(Abf, W3, Hb, Mtiles);
        coef_kernel<<<cdiv((long long)N * Hh, TB), TB, 0, stream>>>(Hb, as3, ad3, es, ed, N, Hh, Cc);
        fill_kernel<<<cdiv((long long)N * Hh, TB), TB, 0, stream>>>(mb, -1e30f, N * Hh);
        fill_kernel<<<cdiv((long long)N * Hh, TB), TB, 0, stream>>>(sb, 0.f, N * Hh);
        fill_kernel<<<cdiv((long long)N * F, TB), TB, 0, stream>>>(O, 0.f, N * F);
        fill_kernel<<<cdiv((long long)G * 16, TB), TB, 0, stream>>>(pool, 0.f, G * 16);
        fill_kernel<<<cdiv((long long)G, TB), TB, 0, stream>>>(cnt, 0.f, G);
        int EH = ET * Hh;
        edge_passA<<<cdiv(EH, TB), TB, 0, stream>>>(ei, E, ET, es, ed, tmp, mb, Hh);
        edge_passB<<<cdiv(EH, TB), TB, 0, stream>>>(ei, E, ET, tmp, mb, sb, Hh);
        edge_passC<<<cdiv(EH, TB), TB, 0, stream>>>(ei, E, ET, tmp, sb, Hb, O, Hh, Cc);
        fin_pool<<<cdiv((long long)N * F, TB), TB, 0, stream>>>(O, b3, bat, pool, N, F);
        count_kernel<<<cdiv((long long)N, TB), TB, 0, stream>>>(bat, cnt, N);
        final_linear<<<1, TB, 0, stream>>>(pool, cnt, Wl, bl, (float*)d_out, G);
    }
}
